// CrfRnnLayerSPIO_80779744903671
// MI455X (gfx1250) — matrix operand order repaired and checked by reading, not compiled, not run on hardware
//
#include <hip/hip_runtime.h>
#include <hip/hip_bf16.h>
#include <stdint.h>

typedef __attribute__((ext_vector_type(16))) _Float16 v16h;
typedef __attribute__((ext_vector_type(8)))  float    v8f;
typedef __attribute__((ext_vector_type(4)))  float    v4f;

#define Hn  384
#define Wn  512
#define Cn  21
#define CP  24                  // padded channel count: 96B/pixel rows, 16B-aligned quads
#define Rn  3
#define HWn (Hn*Wn)
#define TAPS 49
#define TILES_X (Wn/16)
#define TILES_Y (Hn/16)
#define NTILES  (TILES_X*TILES_Y)
#define CSTR 28                 // LDS channel stride (112 B: 16B-aligned, low conflicts)

// dynamic-LDS layout for k_crf_step (bytes)
#define SMT_OFF   0             // 484 pixels * 28ch f32 = 54208, pad to 54272
#define SPATB_OFF 54272         // 16 groups * 32 lanes * 16 elems f16 = 16384
#define BILB_OFF  (SPATB_OFF+16384)
#define BTAB_OFF  (SPATB_OFF+32768) // 105 f32, pad to 448
#define SWG_OFF   (BTAB_OFF+448)    // 49 f32, pad to 256
#define LOWT_OFF  (SWG_OFF+256)     // 21 f32, pad to 96
#define LDS_BYTES (LOWT_OFF+96)

__device__ __constant__ int c_spids[5] = {17, 42, 123, 256, 399};

// ---------------------------------------------------------------- init q[p][24] from unaries[p][21]
__global__ void k_init_q(const float* __restrict__ un, float* __restrict__ qbuf) {
    for (int i = blockIdx.x * blockDim.x + threadIdx.x; i < HWn * CP; i += gridDim.x * blockDim.x) {
        int p = i / CP, c = i - p * CP;
        qbuf[i] = (c < Cn) ? un[p * Cn + c] : 0.f;
    }
}

// ---------------------------------------------------------------- precompute bilateral weights + norms (rgb is iteration-invariant)
__global__ void k_bilat_weights(const float* __restrict__ rgb, float* __restrict__ wbuf,
                                float* __restrict__ snorm, float* __restrict__ bnorm) {
    __shared__ float rgbT[3][22 * 22];
    const int tid = threadIdx.x;
    const int bx = blockIdx.x % TILES_X, by = blockIdx.x / TILES_X;
    const int y0 = by * 16, x0 = bx * 16;
    for (int i = tid; i < 3 * 22 * 22; i += 256) {
        int ch = i / 484, rem = i - ch * 484;
        int hy = rem / 22, hx = rem - hy * 22;
        int gy = y0 - Rn + hy, gx = x0 - Rn + hx;
        float v = 0.f;
        if (gy >= 0 && gy < Hn && gx >= 0 && gx < Wn) v = rgb[(gy * Wn + gx) * 3 + ch];
        rgbT[ch][rem] = v;
    }
    __syncthreads();
    const int ly = tid >> 4, lx = tid & 15;
    const int y = y0 + ly, x = x0 + lx, p = y * Wn + x;
    const int cc = (ly + Rn) * 22 + (lx + Rn);
    const float r0 = rgbT[0][cc], r1 = rgbT[1][cc], r2 = rgbT[2][cc];
    float sn = 0.f, bn = 0.f;
    int tap = 0;
    for (int dy = -Rn; dy <= Rn; ++dy) {
        for (int dx = -Rn; dx <= Rn; ++dx, ++tap) {
            const float d2 = (float)(dy * dy + dx * dx);
            const float swa = __expf(-d2 * (1.f / 51200.f));  // 2*160^2
            const float swg = __expf(-d2 * (1.f / 18.f));     // 2*3^2
            float w = 0.f;
            if ((unsigned)(y + dy) < Hn && (unsigned)(x + dx) < Wn) {
                int nc = (ly + Rn + dy) * 22 + (lx + Rn + dx);
                float e0 = r0 - rgbT[0][nc], e1 = r1 - rgbT[1][nc], e2 = r2 - rgbT[2][nc];
                float cd = e0 * e0 + e1 * e1 + e2 * e2;
                w = swa * __expf(-cd * (1.f / 18.f));
                bn += w;
                sn += swg;
            }
            wbuf[tap * HWn + p] = w;
        }
    }
    snorm[p] = sn;
    bnorm[p] = bn;
}

// ---------------------------------------------------------------- fold matmuls: [Cm@Ws | Cm@Wb] -> f16 WMMA A-fragments
// A-frag layout (16-bit A 16x32, wave32): lane L: m=L%16, g=L/16;
// element e: K = g*16+e. 4 fragments: mt(0,1) x kc(0,1).
__global__ void k_build_mats(const float* __restrict__ Ws, const float* __restrict__ Wb,
                             const float* __restrict__ Cm, _Float16* __restrict__ afrag) {
    __shared__ float Ms[Cn * Cn], Mb[Cn * Cn];
    const int tid = threadIdx.x;
    for (int i = tid; i < Cn * Cn; i += 256) {
        int r = i / Cn, k = i - r * Cn;
        float s = 0.f, b = 0.f;
        for (int j = 0; j < Cn; ++j) {
            float cm = Cm[r * Cn + j];
            s += cm * Ws[j * Cn + k];
            b += cm * Wb[j * Cn + k];
        }
        Ms[i] = s;
        Mb[i] = b;
    }
    __syncthreads();
    for (int i = tid; i < 4 * 32 * 16; i += 256) {
        int frag = i >> 9, lane = (i >> 4) & 31, e = i & 15;
        int mt = frag >> 1, kc = frag & 1;
        int m = lane & 15, g = lane >> 4;
        int K = g * 16 + e;
        int row = mt * 16 + m;
        float v = 0.f;
        if (row < Cn && K < Cn) v = kc ? Mb[row * Cn + K] : Ms[row * Cn + K];
        afrag[i] = (_Float16)v;
    }
}

__global__ void k_zero_bsum(float* __restrict__ bsum) {
    if (threadIdx.x < 128) bsum[threadIdx.x] = 0.f;
}

// ---------------------------------------------------------------- softmax + superpixel log-product reduction
__global__ void k_softmax_sp(const float* __restrict__ qbuf, float* __restrict__ sm,
                             const int* __restrict__ spc, float* __restrict__ bsum) {
    const int p = blockIdx.x * 256 + threadIdx.x;
    const v4f* qp = (const v4f*)(qbuf + (size_t)p * CP);
    float qv[CP];
    v4f q4[6];
#pragma unroll
    for (int j = 0; j < 6; ++j) q4[j] = qp[j];
#pragma unroll
    for (int j = 0; j < 6; ++j)
#pragma unroll
        for (int k = 0; k < 4; ++k) qv[4 * j + k] = q4[j][k];

    float m = -3.4e38f;
#pragma unroll
    for (int c = 0; c < Cn; ++c) m = fmaxf(m, qv[c]);
    float ev[CP], se = 0.f;
#pragma unroll
    for (int c = 0; c < Cn; ++c) { ev[c] = __expf(qv[c] - m); se += ev[c]; }
#pragma unroll
    for (int c = Cn; c < CP; ++c) ev[c] = 0.f;
    const float inv = 1.f / se;
    v4f o4[6];
#pragma unroll
    for (int j = 0; j < 6; ++j)
#pragma unroll
        for (int k = 0; k < 4; ++k) o4[j][k] = ev[4 * j + k] * inv;
    v4f* op = (v4f*)(sm + (size_t)p * CP);
#pragma unroll
    for (int j = 0; j < 6; ++j) op[j] = o4[j];

    const int y = p / Wn, x = p - y * Wn;
    const int sv = spc[x * Hn + y];            // sp_map = transpose(superpixel_cliques)
    int sidx = -1;
#pragma unroll
    for (int s = 0; s < 5; ++s) if (sv == c_spids[s]) sidx = s;
    if (sidx >= 0) {
        const float ls = __logf(se);
#pragma unroll
        for (int c = 0; c < Cn; ++c) atomicAdd(&bsum[sidx * Cn + c], (qv[c] - m) - ls);
    }
}

// ---------------------------------------------------------------- fused: 7x7 filters + WMMA channel mixing + update
__global__ void __launch_bounds__(256)
k_crf_step(const float* __restrict__ sm, const float* __restrict__ wbuf,
           const float* __restrict__ snorm, const float* __restrict__ bnorm,
           const float* __restrict__ bsum, const _Float16* __restrict__ afrag,
           const float* __restrict__ un, const int* __restrict__ spc,
           const float* __restrict__ lowW, const float* __restrict__ highW,
           float* __restrict__ qbuf) {
    extern __shared__ char ldsraw[];
    float*    smT   = (float*)(ldsraw + SMT_OFF);      // sm tile, [pixel][channel] (CSTR pad)
    _Float16* spatB = (_Float16*)(ldsraw + SPATB_OFF); // B-operands in WMMA fragment order
    _Float16* bilB  = (_Float16*)(ldsraw + BILB_OFF);
    float*    Btab  = (float*)(ldsraw + BTAB_OFF);
    float*    swg   = (float*)(ldsraw + SWG_OFF);
    float*    lowT  = (float*)(ldsraw + LOWT_OFF);

    const int tid = threadIdx.x;
    const int bx = blockIdx.x % TILES_X, by = blockIdx.x / TILES_X;
    const int y0 = by * 16, x0 = bx * 16;

    if (tid < TAPS) {
        int dy = tid / 7 - Rn, dx = tid % 7 - Rn;
        swg[tid] = __expf(-(float)(dy * dy + dx * dx) * (1.f / 18.f));
    }
    if (tid < 5 * Cn) Btab[tid] = __expf(bsum[tid]);
    if (tid >= 128 && tid < 128 + Cn) lowT[tid - 128] = lowW[tid - 128];

    // ---- async DMA the sm tile into LDS as 16B quads; per-lane LDS address scatters
    //      the [p][24] rows into the padded [p][28] tile (global side fully coalesced)
    for (int i = tid; i < 484 * 6; i += 256) {
        int rem = i / 6, k = i - rem * 6;              // rem = halo pixel, k = channel quad
        int hy = rem / 22, hx = rem - hy * 22;
        int gy = y0 - Rn + hy, gx = x0 - Rn + hx;
        uint32_t lo = (uint32_t)(uintptr_t)(&smT[rem * CSTR + 4 * k]);  // flat->LDS offset
        if (gy >= 0 && gy < Hn && gx >= 0 && gx < Wn) {
            uint64_t ga = (uint64_t)(uintptr_t)(sm + (size_t)(gy * Wn + gx) * CP + 4 * k);
            asm volatile("global_load_async_to_lds_b128 %0, %1, off"
                         :: "v"(lo), "v"(ga) : "memory");
        } else {
            v4f z = {0.f, 0.f, 0.f, 0.f};
            *(v4f*)(&smT[rem * CSTR + 4 * k]) = z;
        }
    }
    asm volatile("s_wait_asynccnt 0x0" ::: "memory");
    __syncthreads();

    // ---- phase 1: both 7x7 filters; 21 channel reads per tap = 5x ds_load_b128 + 1x b32
    {
        const int ly = tid >> 4, lx = tid & 15;
        const int p = (y0 + ly) * Wn + (x0 + lx);
        float accS[Cn], accB[Cn];
#pragma unroll
        for (int c = 0; c < Cn; ++c) { accS[c] = 0.f; accB[c] = 0.f; }
        int tap = 0;
        for (int dy = -Rn; dy <= Rn; ++dy) {
            for (int dx = -Rn; dx <= Rn; ++dx, ++tap) {
                const float w = wbuf[tap * HWn + p];
                if (tap + 1 < TAPS) __builtin_prefetch(wbuf + (tap + 1) * HWn + p, 0, 0);
                const float sg = swg[tap];
                const float* bp = &smT[((ly + Rn + dy) * 22 + (lx + Rn + dx)) * CSTR];
                v4f vx[5];
#pragma unroll
                for (int j = 0; j < 5; ++j) vx[j] = *(const v4f*)(bp + 4 * j);
                const float v20 = bp[20];
#pragma unroll
                for (int c = 0; c < 20; ++c) {
                    float v = vx[c >> 2][c & 3];
                    accS[c] = fmaf(sg, v, accS[c]);
                    accB[c] = fmaf(w, v, accB[c]);
                }
                accS[20] = fmaf(sg, v20, accS[20]);
                accB[20] = fmaf(w, v20, accB[20]);
            }
        }
        const float invS = 1.f / snorm[p], invB = 1.f / bnorm[p];
        // scatter into WMMA B-fragment order: group=ly, lane=(c/16)*16+lx, elem=c%16
#pragma unroll
        for (int c = 0; c < Cn; ++c) {
            int idx = ((ly * 32 + (c >> 4) * 16 + lx) << 4) + (c & 15);
            spatB[idx] = (_Float16)(accS[c] * invS);
            bilB[idx]  = (_Float16)(accB[c] * invB);
        }
        // zero the unused tail of the K=16..31 chunk (c=21..31)
#pragma unroll
        for (int e = 5; e < 16; ++e) {
            int idx = ((ly * 32 + 16 + lx) << 4) + e;
            spatB[idx] = (_Float16)0.f;
            bilB[idx]  = (_Float16)0.f;
        }
    }
    __syncthreads();

    // ---- phase 2: D(32 x 16px) = [Ms|Mb](32x64,f16) @ [S;B](64x16,f16) via v_wmma_f32_16x16x32_f16
    const v16h* AFv = (const v16h*)afrag;
    const int wv = tid >> 5, lane = tid & 31;
    const int n = lane & 15, g = lane >> 4;
    const float hw = highW[0];
    v16h amk[2][2];                                   // [mt][kc], hoisted out of row loop
#pragma unroll
    for (int mt = 0; mt < 2; ++mt)
#pragma unroll
        for (int kc = 0; kc < 2; ++kc) amk[mt][kc] = AFv[(mt * 2 + kc) * 32 + lane];

    for (int gi = wv * 2; gi < wv * 2 + 2; ++gi) {    // wave owns tile rows gi, gi+1
        v16h b0 = *(const v16h*)(ldsraw + SPATB_OFF + ((gi * 32 + lane) << 5));
        v16h b1 = *(const v16h*)(ldsraw + BILB_OFF  + ((gi * 32 + lane) << 5));

        const int y = y0 + gi, x = x0 + n, p = y * Wn + x;
        const int sv = spc[x * Hn + y];
        int sidx = -1;
#pragma unroll
        for (int s = 0; s < 5; ++s) if (sv == c_spids[s]) sidx = s;
        const int srow = (sidx >= 0 ? sidx : 0) * Cn;
        const float smask = (sidx >= 0) ? 1.f : 0.f;

#pragma unroll
        for (int mt = 0; mt < 2; ++mt) {
            v8f acc = {};
            acc = __builtin_amdgcn_wmma_f32_16x16x32_f16(false, amk[mt][0], false, b0, (short)0, acc, false, false);
            acc = __builtin_amdgcn_wmma_f32_16x16x32_f16(false, amk[mt][1], false, b1, (short)0, acc, false, false);
            // D layout: M = r + 8*g (+16*mt).  mt=0: 8 valid channels (both halves,
            // unconditional, b128-mergeable).  mt=1: 5-channel tail on g=0 half only.
            if (mt == 0 || g == 0) {
                const int nr = (mt == 0) ? 8 : 5;
                const int cb = mt * 16 + 8 * g;
                float* qp = qbuf + (size_t)p * CP + cb;
                const float* up = un + (size_t)p * Cn + cb;
#pragma unroll
                for (int r = 0; r < 8; ++r) {
                    if (r < nr) {
                        float qo = qp[r];
                        float pt = Btab[srow + cb + r] * smask;     // branchless superpixel term
                        float ft = pt * __builtin_amdgcn_rcpf(qo);  // fast rcp (B underflows ~0)
                        float upd = lowT[cb + r] * ft + hw * (1.f - ft);
                        qp[r] = up[r] - acc[r] - upd;
                    }
                }
            }
        }
    }
}

// ---------------------------------------------------------------- final write to (1,H,W,C)
__global__ void k_write_out(const float* __restrict__ qbuf, float* __restrict__ out) {
    for (int i = blockIdx.x * blockDim.x + threadIdx.x; i < HWn * Cn; i += gridDim.x * blockDim.x) {
        int p = i / Cn, c = i - p * Cn;
        out[i] = qbuf[(size_t)p * CP + c];
    }
}

extern "C" void kernel_launch(void* const* d_in, const int* in_sizes, int n_in,
                              void* d_out, int out_size, void* d_ws, size_t ws_size,
                              hipStream_t stream) {
    const float* un   = (const float*)d_in[0];
    const float* rgb  = (const float*)d_in[1];
    const int*   spc  = (const int*)d_in[2];
    const float* Ws   = (const float*)d_in[3];
    const float* Wb   = (const float*)d_in[4];
    const float* low  = (const float*)d_in[5];
    const float* high = (const float*)d_in[6];
    const float* Cm   = (const float*)d_in[7];

    // workspace layout (all L2-resident on MI455X: ~78 MB total vs 192 MB L2)
    char* ws = (char*)d_ws;
    size_t o = 0;
    float* qbuf = (float*)(ws + o); o += (size_t)HWn * CP * 4;     // q, [p][24]
    float* sm   = (float*)(ws + o); o += (size_t)HWn * CP * 4;     // softmax, [p][24]
    float* wbuf = (float*)(ws + o); o += (size_t)TAPS * HWn * 4;   // bilateral weights, iteration-invariant
    float* sn   = (float*)(ws + o); o += (size_t)HWn * 4;
    float* bn   = (float*)(ws + o); o += (size_t)HWn * 4;
    float* bsum = (float*)(ws + o); o += 512;
    _Float16* afrag = (_Float16*)(ws + o);                         // 4 KB, 32B-aligned

    k_init_q<<<4096, 256, 0, stream>>>(un, qbuf);
    k_bilat_weights<<<NTILES, 256, 0, stream>>>(rgb, wbuf, sn, bn);
    k_build_mats<<<1, 256, 0, stream>>>(Ws, Wb, Cm, afrag);

    for (int it = 0; it < 5; ++it) {
        k_zero_bsum<<<1, 128, 0, stream>>>(bsum);
        k_softmax_sp<<<HWn / 256, 256, 0, stream>>>(qbuf, sm, spc, bsum);
        k_crf_step<<<NTILES, 256, LDS_BYTES, stream>>>(sm, wbuf, sn, bn, bsum, afrag,
                                                       un, spc, low, high, qbuf);
    }
    k_write_out<<<4096, 256, 0, stream>>>(qbuf, (float*)d_out);
}
